// MyModel_87522843559492
// MI455X (gfx1250) — matrix operand order repaired and checked by reading, not compiled, not run on hardware
//
#include <hip/hip_runtime.h>

typedef __attribute__((ext_vector_type(16))) _Float16 v16h;
typedef __attribute__((ext_vector_type(8)))  _Float16 v8h;
typedef __attribute__((ext_vector_type(8)))  float    v8f;
typedef __attribute__((ext_vector_type(4)))  float    v4f;

namespace {

constexpr int kT     = 128;   // timesteps
constexpr int kF     = 8;     // input features
constexpr int kTileM = 16;    // batch rows per wave (WMMA M)

// LDS layout (units of _Float16). Row strides are multiples of 8 halves so every
// v8h (ds_load_b128) access is 16B-aligned; +8/+8 padding staggers LDS banks.
constexpr int STR64 = 72;     // stride for K<=64 matrices (144B rows)
constexpr int STR32 = 40;     // stride for K<=32 matrices (80B rows)

constexpr int OFF_WT1 = 0;                       // [128][STR64] K: 0-7 Wk1 | 8-39 Wr1 | 0
constexpr int OFF_WT2 = OFF_WT1 + 128 * STR64;   // [ 64][STR64] K: 0-31 Wk2 | 32-47 Wr2 | 0
constexpr int OFF_WT4 = OFF_WT2 +  64 * STR64;   // [128][STR64] K: 0-31 Wr4 | 32-47 Wk4 | 0
constexpr int OFF_W3R = OFF_WT4 + 128 * STR64;   // [ 64][STR32] K: 0-15 Wr3 | 0
constexpr int OFF_W3K = OFF_W3R +  64 * STR32;   // [ 64][STR32] K: 0-15 Wk3 | 0
constexpr int OFF_WTD = OFF_W3K +  64 * STR32;   // [ 16][STR32] K: 0-31 Wd  (n<5 valid)
constexpr int OFF_SA  = OFF_WTD +  16 * STR32;   // [ 16][STR64] A-operand stage A
constexpr int OFF_SB  = OFF_SA  +  16 * STR64;   // [ 16][STR64] A-operand stage B
constexpr int SMEM_HALVES = OFF_SB + 16 * STR64; // 31104 halves = 62208 B (+1280B logits)

__device__ inline v8f wmma_f16(v16h a, v16h b, v8f c) {
  // v_wmma_f32_16x16x32_f16  (args: neg_a, A, neg_b, B, c_mod, C, reuse_a, reuse_b)
  return __builtin_amdgcn_wmma_f32_16x16x32_f16(false, a, false, b, (short)0, c,
                                                false, false);
}

__device__ inline v8f splat8(float v) {
  v8f r = {v, v, v, v, v, v, v, v};
  return r;
}

// Fast sigmoid: v_exp_f32 + single v_rcp_f32 (no IEEE divide chain on the
// recurrence critical path).
__device__ inline float sigmoidf_fast(float x) {
  return __builtin_amdgcn_rcpf(1.0f + __expf(-x));
}

// A fragment: 16xK f16 tile stored row-major in LDS. Wave32 layout:
// lane l (group g=l>>4) holds row l&15, K = g*8 + kc*32 + {0..7} and +16..+23.
__device__ inline v16h afrag(const _Float16* s, int stride, int lane, int kc) {
  const int r  = lane & 15;
  const int kb = ((lane >> 4) << 3) + (kc << 5);
  v8h lo = *reinterpret_cast<const v8h*>(s + r * stride + kb);
  v8h hi = *reinterpret_cast<const v8h*>(s + r * stride + kb + 16);
  return __builtin_shufflevector(lo, hi, 0, 1, 2, 3, 4, 5, 6, 7,
                                 8, 9, 10, 11, 12, 13, 14, 15);
}

// B fragment from transposed weights WT[n][k] (k contiguous). Wave32 layout:
// lane l holds column n0*16 + (l&15), K = (l>>4)*8 + kc*32 + {0..7} and +16..+23.
__device__ inline v16h bfrag(const _Float16* w, int stride, int lane, int n0, int kc) {
  const int c  = (n0 << 4) + (lane & 15);
  const int kb = ((lane >> 4) << 3) + (kc << 5);
  const v8h* p = reinterpret_cast<const v8h*>(w + c * stride + kb);
  v8h lo = p[0];
  v8h hi = p[2];
  return __builtin_shufflevector(lo, hi, 0, 1, 2, 3, 4, 5, 6, 7,
                                 8, 9, 10, 11, 12, 13, 14, 15);
}

// Scatter one 16-unit half of h (in C/D accumulator layout) into an A-staging tile.
__device__ inline void store_h(_Float16* stage, int lane, int colbase, const float* h8) {
  const int cg = colbase + (lane & 15);
  const int rb = (lane >> 4) << 3;
#pragma unroll
  for (int j = 0; j < 8; ++j) stage[(rb + j) * STR64 + cg] = (_Float16)h8[j];
}

// Keras LSTM cell with activation='relu': i,f,o sigmoid; g relu; h = o*relu(c).
__device__ inline void lstm_gate(const v8f& zi, const v8f& zf, const v8f& zg,
                                 const v8f& zo, v8f& c, float* h8) {
#pragma unroll
  for (int j = 0; j < 8; ++j) {
    float ig = sigmoidf_fast(zi[j]);
    float fg = sigmoidf_fast(zf[j]);
    float gg = fmaxf(zg[j], 0.0f);
    float og = sigmoidf_fast(zo[j]);
    float cn = fg * c[j] + ig * gg;
    c[j]  = cn;
    h8[j] = og * fmaxf(cn, 0.0f);
  }
}

__global__ __launch_bounds__(32) void lstm_autoencoder_fused(
    const float* __restrict__ X,
    const float* __restrict__ Wk1, const float* __restrict__ Wr1, const float* __restrict__ b1,
    const float* __restrict__ Wk2, const float* __restrict__ Wr2, const float* __restrict__ b2,
    const float* __restrict__ Wk3, const float* __restrict__ Wr3, const float* __restrict__ b3,
    const float* __restrict__ Wk4, const float* __restrict__ Wr4, const float* __restrict__ b4,
    const float* __restrict__ Wd,  const float* __restrict__ bd,
    float* __restrict__ out) {
  __shared__ __align__(16) _Float16 smem[SMEM_HALVES];
  __shared__ float slog[16 * 20];

  const int tid  = (int)threadIdx.x;
  const int lane = tid;                 // one wave32 per block
  const int b0   = (int)blockIdx.x * kTileM;
  const int col  = lane & 15;
  const int grp  = lane >> 4;

  // ---------- one-time: stage f16 transposed + K-padded weights in LDS ----------
  for (int i = tid; i < 128 * 64; i += 32) {  // W1 = [Wk1 ; Wr1 ; 0]
    int n = i >> 6, k = i & 63;
    float v = (k < 8) ? Wk1[k * 128 + n] : (k < 40 ? Wr1[(k - 8) * 128 + n] : 0.0f);
    smem[OFF_WT1 + n * STR64 + k] = (_Float16)v;
  }
  for (int i = tid; i < 64 * 64; i += 32) {   // W2 = [Wk2 ; Wr2 ; 0]
    int n = i >> 6, k = i & 63;
    float v = (k < 32) ? Wk2[k * 64 + n] : (k < 48 ? Wr2[(k - 32) * 64 + n] : 0.0f);
    smem[OFF_WT2 + n * STR64 + k] = (_Float16)v;
  }
  for (int i = tid; i < 128 * 64; i += 32) {  // W4 = [Wr4 ; Wk4 ; 0]  (h4 first!)
    int n = i >> 6, k = i & 63;
    float v = (k < 32) ? Wr4[k * 128 + n] : (k < 48 ? Wk4[(k - 32) * 128 + n] : 0.0f);
    smem[OFF_WT4 + n * STR64 + k] = (_Float16)v;
  }
  for (int i = tid; i < 64 * 32; i += 32) {   // Wr3 / Wk3, K padded 16 -> 32
    int n = i >> 5, k = i & 31;
    smem[OFF_W3R + n * STR32 + k] = (_Float16)((k < 16) ? Wr3[k * 64 + n] : 0.0f);
    smem[OFF_W3K + n * STR32 + k] = (_Float16)((k < 16) ? Wk3[k * 64 + n] : 0.0f);
  }
  for (int i = tid; i < 16 * 32; i += 32) {   // Wd, N padded 5 -> 16
    int n = i >> 5, k = i & 31;
    smem[OFF_WTD + n * STR32 + k] = (_Float16)((n < 5) ? Wd[k * 5 + n] : 0.0f);
  }
  for (int i = tid; i < 16 * STR64; i += 32) {  // zero stages: h(-1)=0 and K padding
    smem[OFF_SA + i] = (_Float16)0.0f;
    smem[OFF_SB + i] = (_Float16)0.0f;
  }
  __syncthreads();

  // Per-lane broadcast biases: lane's N-column of each 16-wide output tile.
  float bz1[8], bz4[8], bz2[4], bz3[4];
#pragma unroll
  for (int n = 0; n < 8; ++n) { bz1[n] = b1[n * 16 + col]; bz4[n] = b4[n * 16 + col]; }
#pragma unroll
  for (int n = 0; n < 4; ++n) { bz2[n] = b2[n * 16 + col]; bz3[n] = b3[n * 16 + col]; }
  const float bdv = (col < 5) ? bd[col] : 0.0f;

  _Float16* sA = smem + OFF_SA;
  _Float16* sB = smem + OFF_SB;

  // Hoist phase-A weight B-fragments into VGPRs (loop-invariant): 24 x v16h.
  v16h wB1[8][2], wB2[4][2];
#pragma unroll
  for (int n = 0; n < 8; ++n) {
    wB1[n][0] = bfrag(smem + OFF_WT1, STR64, lane, n, 0);
    wB1[n][1] = bfrag(smem + OFF_WT1, STR64, lane, n, 1);
  }
#pragma unroll
  for (int n = 0; n < 4; ++n) {
    wB2[n][0] = bfrag(smem + OFF_WT2, STR64, lane, n, 0);
    wB2[n][1] = bfrag(smem + OFF_WT2, STR64, lane, n, 1);
  }

  v8f c1a = splat8(0.f), c1b = splat8(0.f), c2 = splat8(0.f);

  // ================= phase A: encoder layers 1+2, fused over t =================
  for (int t = 0; t < kT; ++t) {
    if (lane < 16) {  // x_t -> stage A cols 0..7 (cast to f16)
      const float* xr = X + ((size_t)(b0 + lane) * kT + t) * kF;
      v4f x0 = *reinterpret_cast<const v4f*>(xr);
      v4f x1 = *reinterpret_cast<const v4f*>(xr + 4);
      v8h xh;
#pragma unroll
      for (int e = 0; e < 4; ++e) { xh[e] = (_Float16)x0[e]; xh[4 + e] = (_Float16)x1[e]; }
      *reinterpret_cast<v8h*>(sA + lane * STR64) = xh;
      if (t + 1 < kT) __builtin_prefetch(xr + kF, 0, 3);  // next timestep row
    }
    __syncthreads();

    // layer 1: z1[16,128] = [x_t | h1] @ [Wk1;Wr1] + b1   (K=64 in 2 WMMAs/tile)
    v16h a0 = afrag(sA, STR64, lane, 0);
    v16h a1 = afrag(sA, STR64, lane, 1);
    v8f z1[8];
#pragma unroll
    for (int n = 0; n < 8; ++n) {
      v8f acc = splat8(bz1[n]);
      acc = wmma_f16(a0, wB1[n][0], acc);
      acc = wmma_f16(a1, wB1[n][1], acc);
      z1[n] = acc;
    }
    float h1a[8], h1b[8];
    lstm_gate(z1[0], z1[2], z1[4], z1[6], c1a, h1a);  // units 0..15
    lstm_gate(z1[1], z1[3], z1[5], z1[7], c1b, h1b);  // units 16..31
    __syncthreads();                                   // stage reads done before overwrite
    store_h(sA, lane, 8, h1a);  store_h(sA, lane, 24, h1b);  // h1 -> A cols 8..39
    store_h(sB, lane, 0, h1a);  store_h(sB, lane, 16, h1b);  // h1 -> B cols 0..31
    __syncthreads();

    // layer 2: z2[16,64] = [h1 | h2] @ [Wk2;Wr2] + b2
    v16h a20 = afrag(sB, STR64, lane, 0);
    v16h a21 = afrag(sB, STR64, lane, 1);
    v8f z2[4];
#pragma unroll
    for (int n = 0; n < 4; ++n) {
      v8f acc = splat8(bz2[n]);
      acc = wmma_f16(a20, wB2[n][0], acc);
      acc = wmma_f16(a21, wB2[n][1], acc);
      z2[n] = acc;
    }
    float h2v[8];
    lstm_gate(z2[0], z2[1], z2[2], z2[3], c2, h2v);
    __syncthreads();
    store_h(sB, lane, 32, h2v);                               // h2 -> B cols 32..47
    __syncthreads();
  }

  // ====== bridge (RepeatVector): xz3 = h2_last @ Wk3 + b3, computed ONCE ======
  v8f xz3[4];
  {
    v16h a2h = afrag(sB, STR64, lane, 1);  // h2_last at cols 32..47 (48..63 zero)
#pragma unroll
    for (int n = 0; n < 4; ++n)
      xz3[n] = wmma_f16(a2h, bfrag(smem + OFF_W3K, STR32, lane, n, 0), splat8(bz3[n]));
  }
  __syncthreads();
  for (int i = tid; i < 16 * STR64; i += 32) {  // reset stages: h3(-1)=h4(-1)=0
    sA[i] = (_Float16)0.0f;
    sB[i] = (_Float16)0.0f;
  }
  __syncthreads();

  // Hoist phase-B weight B-fragments into VGPRs (phase-A ones now dead): 21 x v16h.
  v16h wB4[8][2], wB3r[4], wBd;
#pragma unroll
  for (int n = 0; n < 8; ++n) {
    wB4[n][0] = bfrag(smem + OFF_WT4, STR64, lane, n, 0);
    wB4[n][1] = bfrag(smem + OFF_WT4, STR64, lane, n, 1);
  }
#pragma unroll
  for (int n = 0; n < 4; ++n) wB3r[n] = bfrag(smem + OFF_W3R, STR32, lane, n, 0);
  wBd = bfrag(smem + OFF_WTD, STR32, lane, 0, 0);

  v8f c3 = splat8(0.f), c4a = splat8(0.f), c4b = splat8(0.f);

  // ============ phase B: decoder layers 3+4 + dense + softmax, fused ============
  for (int t = 0; t < kT; ++t) {
    // layer 3: z3 = xz3 + h3 @ Wr3   (C-init carries constant input projection)
    v16h a3 = afrag(sA, STR64, lane, 0);
    v8f z3[4];
#pragma unroll
    for (int n = 0; n < 4; ++n) z3[n] = wmma_f16(a3, wB3r[n], xz3[n]);
    float h3v[8];
    lstm_gate(z3[0], z3[1], z3[2], z3[3], c3, h3v);
    __syncthreads();
    store_h(sA, lane, 0, h3v);   // h3 -> A cols 0..15 (layer-3 recurrence)
    store_h(sB, lane, 32, h3v);  // h3 -> B cols 32..47 (layer-4 input)
    __syncthreads();

    // layer 4: z4[16,128] = [h4(t-1) | h3(t)] @ [Wr4;Wk4] + b4
    v16h a40 = afrag(sB, STR64, lane, 0);
    v16h a41 = afrag(sB, STR64, lane, 1);
    v8f z4[8];
#pragma unroll
    for (int n = 0; n < 8; ++n) {
      v8f acc = splat8(bz4[n]);
      acc = wmma_f16(a40, wB4[n][0], acc);
      acc = wmma_f16(a41, wB4[n][1], acc);
      z4[n] = acc;
    }
    float h4a[8], h4b[8];
    lstm_gate(z4[0], z4[2], z4[4], z4[6], c4a, h4a);
    lstm_gate(z4[1], z4[3], z4[5], z4[7], c4b, h4b);
    __syncthreads();
    store_h(sB, lane, 0, h4a);  store_h(sB, lane, 16, h4b);  // h4 -> B cols 0..31
    __syncthreads();

    // dense: logits[16,5] = h4 @ Wd + bd, then per-row softmax
    v16h a_d = afrag(sB, STR64, lane, 0);
    v8f zd = wmma_f16(a_d, wBd, splat8(bdv));
    if (col < 5) {
#pragma unroll
      for (int j = 0; j < 8; ++j) slog[((grp << 3) + j) * 20 + col] = zd[j];
    }
    __syncthreads();
    if (lane < 16) {
      float l0 = slog[lane * 20 + 0], l1 = slog[lane * 20 + 1], l2 = slog[lane * 20 + 2],
            l3 = slog[lane * 20 + 3], l4 = slog[lane * 20 + 4];
      float m  = fmaxf(fmaxf(fmaxf(l0, l1), fmaxf(l2, l3)), l4);
      float e0 = __expf(l0 - m), e1 = __expf(l1 - m), e2 = __expf(l2 - m),
            e3 = __expf(l3 - m), e4 = __expf(l4 - m);
      float rs = __builtin_amdgcn_rcpf(e0 + e1 + e2 + e3 + e4);
      float* o = out + ((size_t)(b0 + lane) * kT + t) * 5;
      o[0] = e0 * rs; o[1] = e1 * rs; o[2] = e2 * rs; o[3] = e3 * rs; o[4] = e4 * rs;
    }
    __syncthreads();
  }
}

}  // namespace

extern "C" void kernel_launch(void* const* d_in, const int* in_sizes, int n_in,
                              void* d_out, int out_size, void* d_ws, size_t ws_size,
                              hipStream_t stream) {
  (void)in_sizes; (void)n_in; (void)out_size; (void)d_ws; (void)ws_size;
  const float* X   = (const float*)d_in[0];
  const float* Wk1 = (const float*)d_in[1];
  const float* Wr1 = (const float*)d_in[2];
  const float* b1  = (const float*)d_in[3];
  const float* Wk2 = (const float*)d_in[4];
  const float* Wr2 = (const float*)d_in[5];
  const float* b2  = (const float*)d_in[6];
  const float* Wk3 = (const float*)d_in[7];
  const float* Wr3 = (const float*)d_in[8];
  const float* b3  = (const float*)d_in[9];
  const float* Wk4 = (const float*)d_in[10];
  const float* Wr4 = (const float*)d_in[11];
  const float* b4  = (const float*)d_in[12];
  const float* Wd  = (const float*)d_in[13];
  const float* bd  = (const float*)d_in[14];
  float* out = (float*)d_out;

  dim3 grid(2048 / kTileM);  // 128 independent recurrent chains, one wave each
  dim3 block(32);
  lstm_autoencoder_fused<<<grid, block, 0, stream>>>(
      X, Wk1, Wr1, b1, Wk2, Wr2, b2, Wk3, Wr3, b3, Wk4, Wr4, b4, Wd, bd, out);
}
